// HiPoNet_41480794145241
// MI455X (gfx1250) — compile-verified, matrix operand order checked by reading, not run on hardware
//
#include <hip/hip_runtime.h>

#define NPAIR 16      // B(4) * n_w(4)
#define NW    4
#define NPTS  2048
#define DIM   32
#define NTILE (NPTS/16)   // 128

// diffusion K-chunking for TDM staging
#define KCHUNK 128                 // K halfs staged per chunk (256B per row)
#define NCHUNK (NPTS / KCHUNK)     // 16 chunks
#define ROWP   (KCHUNK + 8)        // padded LDS row stride in halfs (272B, 4-bank skew)

typedef __attribute__((ext_vector_type(16))) __bf16 v16bf;
typedef __attribute__((ext_vector_type(8)))  __bf16 v8bf;
typedef __attribute__((ext_vector_type(8)))  float  v8f;
typedef unsigned int u32x4 __attribute__((ext_vector_type(4)));
typedef int          i32x4 __attribute__((ext_vector_type(4)));
typedef int          i32x8 __attribute__((ext_vector_type(8)));

// Load one 16-element bf16 operand slice for WMMA A/B per the CDNA5 16-bit
// layout (§7.12.2): lane-half 0 holds K {0..7, 16..23}, half 1 holds {8..15, 24..31},
// i.e. two contiguous 16-byte chunks from a row-major row (global or LDS).
__device__ __forceinline__ v16bf load_row16(const __bf16* row, int half) {
  v8bf lo = *(const v8bf*)(row + half * 8);
  v8bf hi = *(const v8bf*)(row + 16 + half * 8);
  v16bf r;
#pragma unroll
  for (int i = 0; i < 8; ++i) { r[i] = lo[i]; r[i + 8] = hi[i]; }
  return r;
}

// --- 1. Xb = pc[b] * alpha[w]; keep f32 + bf16 copies and row norms ---------
__global__ void prep_kernel(const float* __restrict__ pc,
                            const float* __restrict__ alphas,
                            float* __restrict__ Xf, __bf16* __restrict__ Xh,
                            float* __restrict__ sq) {
  int idx = blockIdx.x * blockDim.x + threadIdx.x;      // p*NPTS + n
  if (idx >= NPAIR * NPTS) return;
  int p = idx / NPTS, n = idx % NPTS;
  int b = p / NW, w = p % NW;
  const float* src = pc + ((size_t)b * NPTS + n) * DIM;
  const float* al  = alphas + w * DIM;
  float s = 0.f;
  size_t o = (size_t)idx * DIM;
#pragma unroll
  for (int d = 0; d < DIM; ++d) {
    float v = src[d] * al[d];
    Xf[o + d] = v;
    Xh[o + d] = (__bf16)v;
    s += v * v;
  }
  sq[idx] = s;
}

// --- 2. Gram via WMMA (NT: both operands are Xh rows), kernel+threshold -----
__global__ void gram_kernel(const __bf16* __restrict__ Xh,
                            const float* __restrict__ sq,
                            const float* __restrict__ sigma_p,
                            __bf16* __restrict__ Wm) {
  int wave = (blockIdx.x * blockDim.x + threadIdx.x) >> 5;
  int lane = threadIdx.x & 31;
  int p  = wave / (NTILE * NTILE);
  int rm = wave % (NTILE * NTILE);
  int i0 = (rm / NTILE) * 16;
  int j0 = (rm % NTILE) * 16;
  int half = lane >> 4, ln = lane & 15;

  float inv_sigma = 1.0f / sigma_p[0];
  const __bf16* Arow = Xh + ((size_t)p * NPTS + i0 + ln) * DIM;
  const __bf16* Brow = Xh + ((size_t)p * NPTS + j0 + ln) * DIM;
  v16bf a  = load_row16(Arow, half);
  v16bf bm = load_row16(Brow, half);
  v8f c = {};
  c = __builtin_amdgcn_wmma_f32_16x16x32_bf16(false, a, false, bm,
                                              (short)0, c, false, false);
  const float* sqp = sq + p * NPTS;
  __bf16* W = Wm + (size_t)p * NPTS * NPTS;
#pragma unroll
  for (int r = 0; r < 8; ++r) {
    int i = i0 + 8 * half + r;
    int j = j0 + ln;
    float dist = sqp[i] + sqp[j] - 2.0f * c[r];
    float wv = __expf(-dist * inv_sigma);
    wv = (wv < 0.5f) ? 0.0f : wv;
    W[(size_t)i * NPTS + j] = (__bf16)wv;
  }
}

// --- 3. column sums -> rcol = 0.5/col (folds the lazy-walk 0.5 factor) -----
__global__ void colsum_kernel(const __bf16* __restrict__ Wm,
                              float* __restrict__ rcol) {
  int p  = blockIdx.x / (NPTS / 256);
  int j  = (blockIdx.x % (NPTS / 256)) * 256 + threadIdx.x;
  const __bf16* W = Wm + (size_t)p * NPTS * NPTS + j;
  float s = 0.f;
  for (int i = 0; i < NPTS; ++i) s += (float)W[(size_t)i * NPTS];
  rcol[p * NPTS + j] = 0.5f / s;   // diag entry is 1.0 -> never zero
}

// --- 4. xs^T[d][n] = bf16(x[n][d] * rcol[n])  (transposed for B-operand) ---
__global__ void scale_kernel(const float* __restrict__ x,
                             const float* __restrict__ rcol,
                             __bf16* __restrict__ xsT) {
  int idx = blockIdx.x * blockDim.x + threadIdx.x;  // (p*DIM+d)*NPTS + n
  if (idx >= NPAIR * DIM * NPTS) return;
  int n  = idx % NPTS;
  int pd = idx / NPTS;
  int d  = pd % DIM, p = pd / DIM;
  xsT[idx] = (__bf16)(x[((size_t)p * NPTS + n) * DIM + d] * rcol[p * NPTS + n]);
}

// --- 5. one lazy-walk step: y = Wm @ xs + 0.5*x -----------------------------
// A-operand (Wm rows, streamed once per step) is staged into LDS by the
// Tensor Data Mover: one tensor_load_to_lds per 16x128 tile chunk, TENSORcnt
// double-buffered. TDM applies the 16B-per-row LDS padding in hardware
// (pad_interval=5 -> every 64 DWORDs, pad_amount=3 -> 4 DWORDs), reproducing
// the 272B bank-skewed row stride for conflict-free WMMA fragment reads.
// One wave per block so the D# descriptor operands are uniform (SGPR-clean).
__global__ void __launch_bounds__(32) diffuse_kernel(
    const __bf16* __restrict__ Wm, const __bf16* __restrict__ xsT,
    const float* __restrict__ xin, float* __restrict__ xout) {
  __shared__ __align__(16) __bf16 stage[2][16 * ROWP];   // 8.7 KB

  int lane = threadIdx.x;
  int wave = blockIdx.x;
  int p  = wave / NTILE;
  int i0 = (wave % NTILE) * 16;
  int half = lane >> 4, ln = lane & 15;

  const __bf16* WmP = Wm + (size_t)p * NPTS * NPTS;
  const __bf16* B0  = xsT + ((size_t)p * DIM + ln) * NPTS;
  const __bf16* B1  = xsT + ((size_t)p * DIM + 16 + ln) * NPTS;

  // D# group 1 (constant across chunks), per ISA §8.4:
  //  data_size=1 (2B), pad_enable=1, pad_interval=5 (64 DW), pad_amount=3 (4 DW)
  //  tensor_dim0 = tensor_dim1 = 2048; tile_dim0 = 128; tile_dim1 = 16
  //  tensor_dim0_stride = 2048
  const i32x8 g1 = { (int)0x07510000,   // mask=0 | data_size<<16 | pad cfg
                     (int)0x08000000,   // tensor_dim0[15:0] @ bits 63:48
                     (int)0x08000000,   // tensor_dim1[15:0] @ bits 95:80
                     (int)0x00800000,   // tile_dim0=128 @ bits 127:112
                     (int)0x00000010,   // tile_dim1=16 @ bits 143:128
                     (int)0x00000800,   // tensor_dim0_stride=2048 @ bits 191:160
                     0, 0 };
  const i32x4 gz = { 0, 0, 0, 0 };      // 2-D tensor: groups 2/3 unused

  auto issue_tdm = [&](int c, int buf) {
    unsigned long long ga = (unsigned long long)(uintptr_t)
        (WmP + (size_t)i0 * NPTS + (size_t)c * KCHUNK);
    unsigned la = (unsigned)(uintptr_t)&stage[buf][0];
    u32x4 g0;
    g0[0] = 1u;                                   // count=1 (valid descriptor)
    g0[1] = la;                                   // lds_addr
    g0[2] = (unsigned)(ga & 0xffffffffu);         // global_addr[31:0]
    g0[3] = (unsigned)(ga >> 32) | 0x80000000u;   // global_addr[56:32] | type=2
#if __clang_major__ >= 23
    __builtin_amdgcn_tensor_load_to_lds(g0, g1, gz, gz,
                                        (i32x8){0,0,0,0,0,0,0,0}, 0);
#else
    __builtin_amdgcn_tensor_load_to_lds(g0, g1, gz, gz, 0);
#endif
  };

  v8f c0 = {}, c1 = {};
  issue_tdm(0, 0);
  for (int c = 0; c < NCHUNK; ++c) {
    int buf = c & 1;
    if (c + 1 < NCHUNK) {
      issue_tdm(c + 1, buf ^ 1);                  // prefetch next chunk via TDM
      __builtin_amdgcn_s_wait_tensorcnt(1);       // in-order: chunk c landed
    } else {
      __builtin_amdgcn_s_wait_tensorcnt(0);
    }
    asm volatile("" ::: "memory");                // keep ds reads below the wait
    const __bf16* Ab = &stage[buf][0];
#pragma unroll
    for (int kk = 0; kk < KCHUNK; kk += 32) {
      int k0 = c * KCHUNK + kk;
      __builtin_prefetch(B0 + k0 + 512, 0, 0);    // keep B stream L2-warm
      v16bf a  = load_row16(Ab + ln * ROWP + kk, half);   // ds_load from LDS
      v16bf b0 = load_row16(B0 + k0, half);
      v16bf b1 = load_row16(B1 + k0, half);
      c0 = __builtin_amdgcn_wmma_f32_16x16x32_bf16(false, a, false, b0,
                                                   (short)0, c0, false, false);
      c1 = __builtin_amdgcn_wmma_f32_16x16x32_bf16(false, a, false, b1,
                                                   (short)0, c1, false, false);
    }
  }
#pragma unroll
  for (int r = 0; r < 8; ++r) {
    int i = i0 + 8 * half + r;
    size_t o = ((size_t)p * NPTS + i) * DIM;
    xout[o + ln]      = c0[r] + 0.5f * xin[o + ln];
    xout[o + 16 + ln] = c1[r] + 0.5f * xin[o + 16 + ln];
  }
}

// --- 6. mean pool over N into output slot ----------------------------------
__global__ void pool_kernel(const float* __restrict__ src,
                            float* __restrict__ out, int slot) {
  __shared__ float red[256];
  int p = blockIdx.x, tid = threadIdx.x;
  int d = tid & 31, g = tid >> 5;                 // 8 row-groups
  float s = 0.f;
  for (int n = g; n < NPTS; n += 8)
    s += src[((size_t)p * NPTS + n) * DIM + d];
  red[tid] = s;
  __syncthreads();
  if (g == 0) {
    float t = 0.f;
#pragma unroll
    for (int gg = 0; gg < 8; ++gg) t += red[gg * 32 + d];
    int b = p / NW, w = p % NW;
    out[(size_t)b * (NW * 5 * DIM) + w * (5 * DIM) + slot * DIM + d] =
        t * (1.0f / NPTS);
  }
}

extern "C" void kernel_launch(void* const* d_in, const int* in_sizes, int n_in,
                              void* d_out, int out_size, void* d_ws, size_t ws_size,
                              hipStream_t stream) {
  (void)in_sizes; (void)n_in; (void)out_size; (void)ws_size;
  const float* pc     = (const float*)d_in[0];   // [4,2048,32]
  const float* sigma  = (const float*)d_in[1];   // scalar
  const float* alphas = (const float*)d_in[2];   // [4,32]
  float* out = (float*)d_out;                    // [4, 640]

  // workspace carve (~144 MB total, dominated by 128 MB bf16 Wm -> L2-resident)
  float*  Xf   = (float*)d_ws;                                  // 4 MB
  __bf16* Xh   = (__bf16*)(Xf + (size_t)NPAIR * NPTS * DIM);    // 2 MB
  float*  sq   = (float*)(Xh + (size_t)NPAIR * NPTS * DIM);     // 128 KB
  float*  rcol = sq + NPAIR * NPTS;                             // 128 KB
  float*  xA   = rcol + NPAIR * NPTS;                           // 4 MB
  float*  xB   = xA + (size_t)NPAIR * NPTS * DIM;               // 4 MB
  __bf16* xsT  = (__bf16*)(xB + (size_t)NPAIR * NPTS * DIM);    // 2 MB
  __bf16* Wm   = (__bf16*)(xsT + (size_t)NPAIR * DIM * NPTS);   // 128 MB

  prep_kernel<<<(NPAIR * NPTS + 255) / 256, 256, 0, stream>>>(pc, alphas, Xf, Xh, sq);
  gram_kernel<<<(NPAIR * NTILE * NTILE) / 8, 256, 0, stream>>>(Xh, sq, sigma, Wm);
  colsum_kernel<<<NPAIR * (NPTS / 256), 256, 0, stream>>>(Wm, rcol);
  pool_kernel<<<NPAIR, 256, 0, stream>>>(Xf, out, 0);          // slot 0: X itself

  int slot = 1;
  for (int t = 1; t <= 8; ++t) {
    const float* xin = (t == 1) ? Xf : ((t & 1) ? xB : xA);
    float* xout = (t & 1) ? xA : xB;
    scale_kernel<<<(NPAIR * DIM * NPTS + 255) / 256, 256, 0, stream>>>(xin, rcol, xsT);
    diffuse_kernel<<<NPAIR * NTILE, 32, 0, stream>>>(Wm, xsT, xin, xout);
    if (t == 1 || t == 2 || t == 4 || t == 8) {
      pool_kernel<<<NPAIR, 256, 0, stream>>>(xout, out, slot);
      ++slot;
    }
  }
}
